// Model_45509473468775
// MI455X (gfx1250) — 8-bit matrix operand order repaired and checked by reading, compiled, not run on hardware
//
#include <hip/hip_runtime.h>

typedef __bf16 bf16;
typedef __attribute__((ext_vector_type(8))) int   v8i;
typedef __attribute__((ext_vector_type(8))) float v8f;

#define T_STEPS 4096
#define BATCH   16
#define ISZ     64
#define HSZ     256
#define NCHUNK  5
#define NLDSCH  2
#define A_STRIDE 336

__device__ __forceinline__ unsigned f32_to_e4m3(float x) {
    unsigned u    = __float_as_uint(x);
    unsigned sign = (u >> 24) & 0x80u;
    unsigned a    = u & 0x7FFFFFFFu;
    a = (a > 0x43E00000u) ? 0x43E00000u : a;
    unsigned r   = a + 0x0007FFFFu + ((a >> 20) & 1u);
    unsigned nrm = (((r >> 23) - 120u) << 3) | ((r >> 20) & 7u);
    unsigned den = (unsigned)__float2int_rn(__uint_as_float(a) * 512.0f);
    return sign | ((a < 0x3C800000u) ? den : nrm);
}

__device__ __forceinline__ unsigned pack4_e4m3(const float* p) {
    return  f32_to_e4m3(p[0])        | (f32_to_e4m3(p[1]) << 8) |
           (f32_to_e4m3(p[2]) << 16) | (f32_to_e4m3(p[3]) << 24);
}

__device__ __forceinline__ float fast_rcp(float v) { return __builtin_amdgcn_rcpf(v); }
__device__ __forceinline__ float fast_sigmoid(float v) {
    return fast_rcp(1.0f + __expf(-v));
}
__device__ __forceinline__ float fast_tanh(float v) {
    return __fmaf_rn(fast_rcp(1.0f + __expf(-2.0f * v)), 2.0f, -1.0f);
}

__launch_bounds__(1024)
__global__ void lstm_persistent_fp8(const float* __restrict__ x,
                                    const float* __restrict__ W_ih,
                                    const float* __restrict__ W_hh,
                                    const float* __restrict__ b_ih,
                                    const float* __restrict__ b_hh,
                                    const float* __restrict__ W_lin,
                                    const float* __restrict__ b_lin,
                                    const float* __restrict__ h0,
                                    const float* __restrict__ c0,
                                    float* __restrict__ out)
{
    __shared__ unsigned char A_sh[16 * A_STRIDE];
    __shared__ unsigned char W01_sh[NLDSCH * 64 * 32 * 32];
    __shared__ bf16  G_sh[64 * 256];
    __shared__ float C_sh[16 * 32 * 8];
    __shared__ float H_fin[16 * 256];

    const int tid  = threadIdx.x;
    const int lane = tid & 31;
    const int wv   = tid >> 5;
    const int hf   = lane >> 4;
    const int ln   = lane & 15;

    v8i  wreg[2][NCHUNK - NLDSCH];
    float bias[2];
    #pragma unroll
    for (int tt = 0; tt < 2; ++tt) {
        const int g = (2 * wv + tt) * 16 + ln;
        bias[tt] = b_ih[g] + b_hh[g];
        #pragma unroll
        for (int c = 0; c < NCHUNK; ++c) {
            const float* src = (c == 0) ? (W_ih + g * ISZ)
                                        : (W_hh + g * HSZ + (c - 1) * 64);
            const int kb0 = hf * 16;
            v8i w;
            #pragma unroll
            for (int seg = 0; seg < 2; ++seg) {
                const float* p = src + kb0 + seg * 32;
                #pragma unroll
                for (int d = 0; d < 4; ++d)
                    w[seg * 4 + d] = (int)pack4_e4m3(p + d * 4);
            }
            if (c < NLDSCH)
                *(v8i*)&W01_sh[((c * 64 + 2 * wv + tt) * 32 + lane) * 32] = w;
            else
                wreg[tt][c - NLDSCH] = w;
        }
    }

    if (wv >= 16) {
        const int lid = (wv - 16) * 32 + lane;
        #pragma unroll
        for (int s = 0; s < 2; ++s) {
            const int e = lid + s * 512;
            const int b = e >> 6, k = e & 63;
            A_sh[b * A_STRIDE + k] = (unsigned char)f32_to_e4m3(x[b * ISZ + k]);
        }
    } else {
        const int j = wv;
        float* cp = &C_sh[(j * 32 + lane) * 8];
        #pragma unroll
        for (int r = 0; r < 8; ++r) {
            const int b = r + hf * 8;
            const int n = 16 * j + ln;
            A_sh[b * A_STRIDE + ISZ + n] =
                (unsigned char)f32_to_e4m3(h0[b * HSZ + n]);
            cp[r] = c0[b * HSZ + n];
        }
    }
    __syncthreads();

    for (int t = 0; t < T_STEPS; ++t) {
        v8f acc0, acc1;
        #pragma unroll
        for (int r = 0; r < 8; ++r) { acc0[r] = bias[0]; acc1[r] = bias[1]; }

        #pragma unroll
        for (int c = 0; c < NCHUNK; ++c) {
            const unsigned char* row = &A_sh[ln * A_STRIDE + c * 64 + hf * 16];
            v8i a;
            #pragma unroll
            for (int s = 0; s < 4; ++s) {
                const unsigned long long q =
                    *(const unsigned long long*)(row + (s >> 1) * 32 + (s & 1) * 8);
                a[2 * s]     = (int)(unsigned)q;
                a[2 * s + 1] = (int)(unsigned)(q >> 32);
            }
            v8i b0, b1;
            if (c < NLDSCH) {
                b0 = *(const v8i*)&W01_sh[((c * 64 + 2 * wv + 0) * 32 + lane) * 32];
                b1 = *(const v8i*)&W01_sh[((c * 64 + 2 * wv + 1) * 32 + lane) * 32];
            } else {
                b0 = wreg[0][c - NLDSCH];
                b1 = wreg[1][c - NLDSCH];
            }
            acc0 = __builtin_amdgcn_wmma_f32_16x16x64_fp8_fp8(
                       a, b0, (short)0, acc0, false, false);
            acc1 = __builtin_amdgcn_wmma_f32_16x16x64_fp8_fp8(
                       a, b1, (short)0, acc1, false, false);
        }

        const bool  isG  = (wv >> 3) == 2;
        const float sIn  = isG ? 2.0f : 1.0f;
        const float sMul = isG ? 2.0f : 1.0f;
        const float sAdd = isG ? -1.0f : 0.0f;
        #pragma unroll
        for (int tt = 0; tt < 2; ++tt) {
            const v8f acc = tt ? acc1 : acc0;
            bf16* gb = &G_sh[(2 * wv + tt) * 256];
            #pragma unroll
            for (int r = 0; r < 8; ++r) {
                const float s = fast_rcp(1.0f + __expf(-sIn * acc[r]));
                gb[r * 32 + hf * 16 + ln] = (bf16)__fmaf_rn(s, sMul, sAdd);
            }
        }
        __syncthreads();

        if (wv < 16) {
            const int j = wv;
            const bf16* gi = &G_sh[(j)      * 256];
            const bf16* gf = &G_sh[(16 + j) * 256];
            const bf16* gg = &G_sh[(32 + j) * 256];
            const bf16* go = &G_sh[(48 + j) * 256];
            float* cp = &C_sh[(j * 32 + lane) * 8];
            #pragma unroll
            for (int r = 0; r < 8; ++r) {
                const int off = r * 32 + hf * 16 + ln;
                const float cv = (float)gf[off] * cp[r] +
                                 (float)gi[off] * (float)gg[off];
                cp[r] = cv;
                const float hv = (float)go[off] * fast_tanh(cv);
                const int b = r + hf * 8;
                const int n = 16 * j + ln;
                A_sh[b * A_STRIDE + ISZ + n] = (unsigned char)f32_to_e4m3(hv);
                if (t == T_STEPS - 1) H_fin[b * HSZ + n] = hv;
            }
        } else {
            if (t + 2 < T_STEPS && wv == 16) {
                __builtin_prefetch(x + (size_t)(t + 2) * (BATCH * ISZ) + lane * 32, 0, 1);
            }
            if (t + 1 < T_STEPS) {
                const int lid = (wv - 16) * 32 + lane;
                const float* xt = x + (size_t)(t + 1) * (BATCH * ISZ);
                #pragma unroll
                for (int s = 0; s < 2; ++s) {
                    const int e = lid + s * 512;
                    const int b = e >> 6, k = e & 63;
                    A_sh[b * A_STRIDE + k] =
                        (unsigned char)f32_to_e4m3(xt[b * ISZ + k]);
                }
            }
        }
        __syncthreads();
    }

    if (tid < BATCH) {
        const int b = tid;
        float sum = b_lin[0];
        #pragma unroll 8
        for (int n = 0; n < HSZ; ++n)
            sum += H_fin[b * HSZ + n] * W_lin[n];
        out[b] = fast_sigmoid(sum);
    }
}

extern "C" void kernel_launch(void* const* d_in, const int* in_sizes, int n_in,
                              void* d_out, int out_size, void* d_ws, size_t ws_size,
                              hipStream_t stream) {
    (void)in_sizes; (void)n_in; (void)d_ws; (void)ws_size; (void)out_size;
    const float* x     = (const float*)d_in[0];
    const float* W_ih  = (const float*)d_in[1];
    const float* W_hh  = (const float*)d_in[2];
    const float* b_ih  = (const float*)d_in[3];
    const float* b_hh  = (const float*)d_in[4];
    const float* W_lin = (const float*)d_in[5];
    const float* b_lin = (const float*)d_in[6];
    const float* h0    = (const float*)d_in[7];
    const float* c0    = (const float*)d_in[8];
    lstm_persistent_fp8<<<1, 1024, 0, stream>>>(
        x, W_ih, W_hh, b_ih, b_hh, W_lin, b_lin, h0, c0, (float*)d_out);
}
